// LabeledBinaryTreeGRU_63857573757105
// MI455X (gfx1250) — matrix operand order repaired and checked by reading, not compiled, not run on hardware
//
#include <hip/hip_runtime.h>
#include <stdint.h>
#include <stddef.h>

typedef __bf16 bf16;
typedef __attribute__((ext_vector_type(16))) __bf16 v16bf;
typedef __attribute__((ext_vector_type(8)))  __bf16 v8bf;
typedef __attribute__((ext_vector_type(8)))  float  v8f;

// Problem constants
constexpr int kH      = 300;    // hidden dim / N
constexpr int kNLeaf  = 4096;
constexpr int kNNodes = 8191;

// Tiling constants
constexpr int kXS   = 992;      // padded LDS x-row stride (bf16 elements), 31*32
constexpr int kRS   = 304;      // padded N stride in LDS (19 tiles * 16)
constexpr int kNTP  = 24;       // padded N-tile count: 8 waves x 3 tiles (19 real)
constexpr int kKCRZ = 31;       // K chunks of 32 for r/z GEMM (covers cols 0..991)
constexpr int kKCU  = 29;       // K chunks of 32 for u GEMM (covers cols 64..991)

// ---------------------------------------------------------------------------
// Weight swizzle: f32 (K x 300, row-major) -> bf16 B-fragments.
// r/z fragments interleaved: for fragment (kc,nt),
//   Wr frag at ((kc*kNTP+nt)*2 + 0)*512,  Wz frag at ((kc*kNTP+nt)*2 + 1)*512
// Within a fragment: [lane*16 + j] = W'[32*kc + 16*(lane>>4) + j][nt*16 + (lane&15)]
// in contiguous k order, the same order load_a_frag uses (lanes 0-15: K=0..15,
// lanes 16-31: K=16..31 of each 32-deep chunk). Tiles nt>=19 are zero pads.
// K permuted to the LDS x layout: te@[0,50) we@[64,364) lh@[364,664) rh@[664,964).
// Original Wr/Wz rows: we[0,300) te[300,350) lh[350,650) rh[650,950).
// ---------------------------------------------------------------------------
__global__ void conv_rz_kernel(const float* __restrict__ Wr,
                               const float* __restrict__ Wz,
                               bf16* __restrict__ dst) {
  int idx = blockIdx.x * 256 + threadIdx.x;
  const int total = kKCRZ * kNTP * 2 * 512;
  if (idx >= total) return;
  int j    = idx & 15;
  int lane = (idx >> 4) & 31;
  int mat  = (idx >> 9) & 1;
  int frag = idx >> 10;
  int nt = frag % kNTP, kc = frag / kNTP;
  int kp = 32 * kc + 16 * (lane >> 4) + j;
  int n  = nt * 16 + (lane & 15);
  int orig = -1;
  if (kp < 50)                    orig = 300 + kp;   // te
  else if (kp >= 64 && kp < 364)  orig = kp - 64;    // we
  else if (kp >= 364 && kp < 964) orig = kp - 14;    // lh/rh (orig 350..949)
  float v = 0.f;
  if (orig >= 0 && n < kH) v = (mat ? Wz : Wr)[orig * kH + n];
  dst[idx] = (bf16)v;
}

__global__ void conv_u_kernel(const float* __restrict__ W, bf16* __restrict__ dst) {
  int idx = blockIdx.x * 256 + threadIdx.x;
  const int total = kKCU * kNTP * 512;
  if (idx >= total) return;
  int j    = idx & 15;
  int lane = (idx >> 4) & 31;
  int frag = idx >> 9;
  int nt = frag % kNTP, kc = frag / kNTP;
  int kp = 32 * kc + 16 * (lane >> 4) + j;   // u K-window: x cols 64..991 -> k'' 0..927
  int n  = nt * 16 + (lane & 15);
  float v = 0.f;
  if (kp < 900 && n < kH) v = W[kp * kH + n]; // Wu rows: we[0,300) rlh[300,600) rrh[600,900)
  dst[idx] = (bf16)v;
}

// ---------------------------------------------------------------------------
// A-fragment load from LDS in contiguous k order (same as the B fragments):
// lane l: M = l&15; lanes 0-15 hold K = base+0..15,
//         lanes 16-31 hold K = base+16..31.
// Two 16-byte ds_load_b128s per fragment (all offsets 16B aligned).
// ---------------------------------------------------------------------------
__device__ __forceinline__ v16bf load_a_frag(const bf16* __restrict__ xbuf,
                                             int lane, int colbase) {
  int mrow = lane & 15, half = lane >> 4;
  const bf16* p = xbuf + mrow * kXS + colbase + 16 * half;
  v8bf lo = *(const v8bf*)p;
  v8bf hi = *(const v8bf*)(p + 8);
  return __builtin_shufflevector(lo, hi, 0,1,2,3,4,5,6,7,8,9,10,11,12,13,14,15);
}

__device__ __forceinline__ float sigmoidf_(float x) {
  return 1.f / (1.f + __expf(-x));
}

// ---------------------------------------------------------------------------
// One tree level: each block handles 16 nodes; each of the 8 wave32s owns
// 3 N-tiles (t = wave, wave+8, wave+16; padded tiles are dummies).
// Loop order: kc outer, tiles inner -> each A fragment is loaded from LDS
// once and consumed immediately by all 3 tiles (no long live ranges, no
// spills). __launch_bounds__(256, 4) keeps the wave under 256 VGPRs.
// ---------------------------------------------------------------------------
__global__ __launch_bounds__(256, 4) void tree_gru_level_kernel(
    const int*   __restrict__ leaf_word_ids,
    const int*   __restrict__ tag_ids,
    const float* __restrict__ word_table,
    const float* __restrict__ tag_table,
    const float* __restrict__ br, const float* __restrict__ bz,
    const float* __restrict__ bu,
    const bf16*  __restrict__ WrzF, const bf16* __restrict__ WuF,
    const float* __restrict__ prev_h,
    float*       __restrict__ out_level,
    float*       __restrict__ out_final,
    int m, int node_off, int is_leaf, int is_last)
{
  __shared__ __align__(16) bf16  xbuf[16 * kXS];   // te|we|lh|rh (later lh*=r, rh*=r)
  __shared__ __align__(16) bf16  rbuf[16 * kRS];   // r gate
  __shared__ __align__(16) float sumbuf[16 * kRS]; // lh + rh (f32)

  const int tid  = threadIdx.x;
  const int blk  = blockIdx.x;
  const int lane = tid & 31;
  const int wave = tid >> 5;

  // ---- Phase A: zero + gather --------------------------------------------
  for (int i = tid; i < 16 * kXS / 2; i += 256) ((uint32_t*)xbuf)[i] = 0u;
  for (int i = tid; i < 16 * kRS; i += 256) sumbuf[i] = 0.f;
  __syncthreads();

  for (int idx = tid; idx < 16 * 50; idx += 256) {          // tag embeddings
    int i = idx / 50, c = idx - i * 50;
    int g = blk * 16 + i;
    if (g < m) {
      int tg = tag_ids[node_off + g];
      xbuf[i * kXS + c] = (bf16)tag_table[tg * 50 + c];
    }
  }
  if (is_leaf) {
    for (int idx = tid; idx < 16 * kH; idx += 256) {        // word embeddings
      int i = idx / kH, c = idx - i * kH;
      int g = blk * 16 + i;
      if (g < m) {
        int w = leaf_word_ids[g];
        xbuf[i * kXS + 64 + c] = (bf16)word_table[(size_t)w * kH + c];
      }
    }
  } else {
    for (int idx = tid; idx < 16 * kH; idx += 256) {        // child states
      int i = idx / kH, c = idx - i * kH;
      int g = blk * 16 + i;
      if (g < m) {
        float lv = prev_h[(size_t)(2 * g) * kH + c];
        float rv = prev_h[(size_t)(2 * g + 1) * kH + c];
        xbuf[i * kXS + 364 + c] = (bf16)lv;
        xbuf[i * kXS + 664 + c] = (bf16)rv;
        sumbuf[i * kRS + c] = lv + rv;
      }
    }
  }
  __syncthreads();

  float zreg[3][8];

  // ---- Phase B: r and z gates (K window = cols 0..991) -------------------
  {
    v8f accr[3] = {{0,0,0,0,0,0,0,0},{0,0,0,0,0,0,0,0},{0,0,0,0,0,0,0,0}};
    v8f accz[3] = {{0,0,0,0,0,0,0,0},{0,0,0,0,0,0,0,0},{0,0,0,0,0,0,0,0}};
    #pragma unroll 2
    for (int kc = 0; kc < kKCRZ; ++kc) {
      v16bf a = load_a_frag(xbuf, lane, 32 * kc);
      const bf16* base = WrzF + (((size_t)kc * kNTP + wave) * 2) * 512 + lane * 16;
      #pragma unroll
      for (int q = 0; q < 3; ++q) {                 // tiles t = wave + 8*q
        const bf16* fp = base + (size_t)q * 8 * 1024;
        v16bf wr = *(const v16bf*)fp;
        v16bf wz = *(const v16bf*)(fp + 512);
        accr[q] = __builtin_amdgcn_wmma_f32_16x16x32_bf16(false, a, false, wr,
                                                          (short)0, accr[q], false, false);
        accz[q] = __builtin_amdgcn_wmma_f32_16x16x32_bf16(false, a, false, wz,
                                                          (short)0, accz[q], false, false);
      }
    }
    #pragma unroll
    for (int q = 0; q < 3; ++q) {
      int t = wave + 8 * q;
      int n = t * 16 + (lane & 15);
      float bR = (n < kH) ? br[n] : 0.f;
      float bZ = (n < kH) ? bz[n] : 0.f;
      #pragma unroll
      for (int v = 0; v < 8; ++v) {
        int mr = (lane >> 4) * 8 + v;               // C/D layout: M = 8*(lane/16)+vgpr
        float rr = sigmoidf_(accr[q][v] + bR);
        if (n < kH) rbuf[mr * kRS + n] = (bf16)rr;
        zreg[q][v] = sigmoidf_(accz[q][v] + bZ);
      }
    }
  }
  __syncthreads();

  // ---- Phase C: in-place X2 = [we | r*lh | r*rh] -------------------------
  for (int idx = tid; idx < 16 * 600; idx += 256) {
    int i = idx / 600, c = idx - i * 600;
    int n = (c < kH) ? c : (c - kH);
    int col = 364 + c;                              // lh cols 364.., rh cols 664..
    float v = (float)xbuf[i * kXS + col] * (float)rbuf[i * kRS + n];
    xbuf[i * kXS + col] = (bf16)v;
  }
  __syncthreads();

  // ---- Phase D: u gate (K window = cols 64..991) and combine -------------
  {
    v8f accu[3] = {{0,0,0,0,0,0,0,0},{0,0,0,0,0,0,0,0},{0,0,0,0,0,0,0,0}};
    #pragma unroll 2
    for (int kc = 0; kc < kKCU; ++kc) {
      v16bf a = load_a_frag(xbuf, lane, 64 + 32 * kc);
      const bf16* base = WuF + ((size_t)kc * kNTP + wave) * 512 + lane * 16;
      #pragma unroll
      for (int q = 0; q < 3; ++q) {
        v16bf wu = *(const v16bf*)(base + (size_t)q * 8 * 512);
        accu[q] = __builtin_amdgcn_wmma_f32_16x16x32_bf16(false, a, false, wu,
                                                          (short)0, accu[q], false, false);
      }
    }
    #pragma unroll
    for (int q = 0; q < 3; ++q) {
      int t = wave + 8 * q;
      int n = t * 16 + (lane & 15);
      float bU = (n < kH) ? bu[n] : 0.f;
      #pragma unroll
      for (int v = 0; v < 8; ++v) {
        int mr = (lane >> 4) * 8 + v;
        float u = tanhf(accu[q][v] + bU);
        float z = zreg[q][v];
        float s = sumbuf[mr * kRS + n < 16 * kRS ? mr * kRS + n : 0];
        float h = z * s + (1.f - z) * u;
        int g = blk * 16 + mr;
        if (g < m && n < kH) {
          out_level[(size_t)g * kH + n] = h;
          if (is_last && g == 0) out_final[n] = h;
        }
      }
    }
  }
}

// ---------------------------------------------------------------------------
extern "C" void kernel_launch(void* const* d_in, const int* in_sizes, int n_in,
                              void* d_out, int out_size, void* d_ws, size_t ws_size,
                              hipStream_t stream) {
  const int*   leaf_word_ids = (const int*)d_in[0];
  const int*   tag_ids       = (const int*)d_in[1];
  const float* word_table    = (const float*)d_in[2];
  const float* tag_table     = (const float*)d_in[3];
  const float* Wr            = (const float*)d_in[4];
  const float* br            = (const float*)d_in[5];
  const float* Wz            = (const float*)d_in[6];
  const float* bz            = (const float*)d_in[7];
  const float* Wu            = (const float*)d_in[8];
  const float* bu            = (const float*)d_in[9];
  float* out = (float*)d_out;

  // Workspace: bf16 fragment-swizzled weights (r/z interleaved, then u)
  char* ws = (char*)d_ws;
  const size_t rz_elems = (size_t)kKCRZ * kNTP * 2 * 512;
  const size_t u_elems  = (size_t)kKCU  * kNTP * 512;
  bf16* WrzF = (bf16*)ws;
  bf16* WuF  = WrzF + rz_elems;

  conv_rz_kernel<<<(int)((rz_elems + 255) / 256), 256, 0, stream>>>(Wr, Wz, WrzF);
  conv_u_kernel<<<(int)((u_elems + 255) / 256), 256, 0, stream>>>(Wu, WuF);

  float* out_final = out + (size_t)kNNodes * kH;

  int m = kNLeaf;
  int node_off = 0;
  int prev_off = 0;
  int level = 0;
  while (m >= 1) {
    int is_leaf = (level == 0);
    int is_last = (m == 1);
    const float* prev_h = is_leaf ? (const float*)nullptr
                                  : out + (size_t)prev_off * kH;
    float* out_level = out + (size_t)node_off * kH;
    int blocks = (m + 15) / 16;
    tree_gru_level_kernel<<<blocks, 256, 0, stream>>>(
        leaf_word_ids, tag_ids, word_table, tag_table, br, bz, bu,
        WrzF, WuF, prev_h, out_level, out_final,
        m, node_off, is_leaf, is_last);
    prev_off = node_off;
    node_off += m;
    m >>= 1;
    ++level;
  }
}
